// MultiHeadAttention_32676111188129
// MI455X (gfx1250) — compile-verified, matrix operand order checked by reading, not run on hardware
//
#include <hip/hip_runtime.h>
#include <hip/hip_bf16.h>

#define S 2048
#define D 1024
#define H 16
#define DK 64
#define BDIM 2

#define BM 128  // block tile rows
#define BN 64   // block tile cols
#define BK 32   // K step
#define SA 40   // LDS row stride in ushorts (80 bytes: 16B aligned, conflict-skewed)

typedef float        v8f   __attribute__((ext_vector_type(8)));
typedef unsigned int v8u   __attribute__((ext_vector_type(8)));
typedef __bf16       v16bf __attribute__((ext_vector_type(16)));
typedef int          v4i_vs __attribute__((__vector_size__(16)));

// Async global->LDS staging (CDNA5 GLOBAL_LOAD_ASYNC_TO_LDS_B128, ASYNCcnt).
// Guarded: falls back to VGPR-bounce copies if the toolchain lacks the builtins.
#if __has_builtin(__builtin_amdgcn_global_load_async_to_lds_b128) && \
    __has_builtin(__builtin_amdgcn_s_wait_asynccnt)
#define ASYNC_COPY 1
#endif

#ifdef ASYNC_COPY
__device__ __forceinline__ void async_cp16(const unsigned short* g,
                                           unsigned short* l) {
  __builtin_amdgcn_global_load_async_to_lds_b128(
      (__attribute__((address_space(1))) v4i_vs*)(void*)g,
      (__attribute__((address_space(3))) v4i_vs*)(void*)l, 0, 0);
}
__device__ __forceinline__ void async_wait() {
  __builtin_amdgcn_s_wait_asynccnt(0);
}
#else
__device__ __forceinline__ void async_cp16(const unsigned short* g,
                                           unsigned short* l) {
  *reinterpret_cast<uint4*>(l) = *reinterpret_cast<const uint4*>(g);
}
__device__ __forceinline__ void async_wait() {}
#endif

__device__ __forceinline__ unsigned short f2bf(float f) {
  unsigned int u = __float_as_uint(f);
  u += 0x7FFFu + ((u >> 16) & 1u);  // round-to-nearest-even
  return (unsigned short)(u >> 16);
}

// Build a 16-element bf16 fragment from two 16B LDS chunks (ds_load_b128 x2).
// CDNA5 16-bit A/B layout: lanes 0-15 take K 0-7 & 16-23 (VGPR 0-3,4-7),
// lanes 16-31 take K 8-15 & 24-31 -- caller passes pointers offset by
// half*8 and 16+half*8.
__device__ __forceinline__ v16bf frag16(const unsigned short* p0,
                                        const unsigned short* p1) {
  uint4 a = *reinterpret_cast<const uint4*>(p0);
  uint4 b = *reinterpret_cast<const uint4*>(p1);
  v8u u = {a.x, a.y, a.z, a.w, b.x, b.y, b.z, b.w};
  return __builtin_bit_cast(v16bf, u);
}

#define WMMA_BF16(A, B, C)                                                  \
  __builtin_amdgcn_wmma_f32_16x16x32_bf16(false, (A), false, (B), (short)0, \
                                          (C), false, false)

// 8 waves as 4x2 grid, each wave owns a 32x32 output tile (4 accumulators,
// 4 WMMAs per K-step from 8 ds_load_b128 fragment loads).
#define WAVE_COORDS()                              \
  const int t = threadIdx.x;                       \
  const int lane = t & 31, wid = t >> 5;           \
  const int wm = wid >> 1, wn = wid & 1;           \
  const int half = lane >> 4, l16 = lane & 15

#define WMMA_STEP(As, Bs, acc)                                              \
  do {                                                                      \
    const int r0_ = wm * 32 + l16;                                          \
    const int c0_ = wn * 32 + l16;                                          \
    v16bf fa0 = frag16(&As[r0_ * SA + half * 8],                            \
                       &As[r0_ * SA + 16 + half * 8]);                      \
    v16bf fa1 = frag16(&As[(r0_ + 16) * SA + half * 8],                     \
                       &As[(r0_ + 16) * SA + 16 + half * 8]);               \
    v16bf fb0 = frag16(&Bs[c0_ * SA + half * 8],                            \
                       &Bs[c0_ * SA + 16 + half * 8]);                      \
    v16bf fb1 = frag16(&Bs[(c0_ + 16) * SA + half * 8],                     \
                       &Bs[(c0_ + 16) * SA + 16 + half * 8]);               \
    acc[0] = WMMA_BF16(fa0, fb0, acc[0]);                                   \
    acc[1] = WMMA_BF16(fa0, fb1, acc[1]);                                   \
    acc[2] = WMMA_BF16(fa1, fb0, acc[2]);                                   \
    acc[3] = WMMA_BF16(fa1, fb1, acc[3]);                                   \
  } while (0)

// ---------------------------------------------------------------------------
// Generic  Out[M,N] = A[M,K] @ W[K,N] + bias   (W fp32, transposed into LDS)
// ---------------------------------------------------------------------------
template <bool A_BF16, bool OUT_F32>
__global__ void gemm_bias_kernel(const void* __restrict__ Ap,
                                 const float* __restrict__ W,
                                 const float* __restrict__ bias,
                                 void* __restrict__ Out,
                                 int M, int N, int K) {
  __shared__ unsigned short As[BM * SA];
  __shared__ unsigned short Bs[BN * SA];
  WAVE_COORDS();

  const int m0 = blockIdx.y * BM;
  const int n0 = blockIdx.x * BN;

  v8f acc[4] = {{0.f, 0.f, 0.f, 0.f, 0.f, 0.f, 0.f, 0.f}};
  acc[1] = acc[0]; acc[2] = acc[0]; acc[3] = acc[0];

  for (int k0 = 0; k0 < K; k0 += BK) {
    // ---- stage A tile (128x32) into LDS as bf16 ----
    if constexpr (A_BF16) {
      const unsigned short* A = (const unsigned short*)Ap;
      int ar = t >> 2, ac = (t & 3) * 8;
#pragma unroll
      for (int it = 0; it < 2; ++it) {
        int r = ar + it * 64;
        async_cp16(&A[(size_t)(m0 + r) * K + k0 + ac], &As[r * SA + ac]);
      }
    } else {
      const float* A = (const float*)Ap;
      int ar = t >> 3, ac = (t & 7) * 4;
#pragma unroll
      for (int it = 0; it < 4; ++it) {
        int r = ar + it * 32;
        float4 f = *reinterpret_cast<const float4*>(
            &A[(size_t)(m0 + r) * K + k0 + ac]);
        unsigned short* d = &As[r * SA + ac];
        d[0] = f2bf(f.x); d[1] = f2bf(f.y); d[2] = f2bf(f.z); d[3] = f2bf(f.w);
      }
    }
    // ---- stage B tile transposed: Bs[n][k] = W[k0+k][n0+n] ----
    {
      int bk = t >> 4, bn = (t & 15) * 4;
#pragma unroll
      for (int it = 0; it < 2; ++it) {
        int k = bk + it * 16;
        float4 f = *reinterpret_cast<const float4*>(
            &W[(size_t)(k0 + k) * N + n0 + bn]);
        Bs[(bn + 0) * SA + k] = f2bf(f.x);
        Bs[(bn + 1) * SA + k] = f2bf(f.y);
        Bs[(bn + 2) * SA + k] = f2bf(f.z);
        Bs[(bn + 3) * SA + k] = f2bf(f.w);
      }
    }
    if (A_BF16) async_wait();
    __syncthreads();
    WMMA_STEP(As, Bs, acc);
    __syncthreads();
  }

#pragma unroll
  for (int ms = 0; ms < 2; ++ms)
#pragma unroll
    for (int ns = 0; ns < 2; ++ns)
#pragma unroll
      for (int r = 0; r < 8; ++r) {
        int gr = m0 + wm * 32 + ms * 16 + half * 8 + r;
        int gc = n0 + wn * 32 + ns * 16 + l16;
        float v = acc[ms * 2 + ns][r] + bias[gc];
        if constexpr (OUT_F32)
          ((float*)Out)[(size_t)gr * N + gc] = v;
        else
          ((unsigned short*)Out)[(size_t)gr * N + gc] = f2bf(v);
      }
}

// ---------------------------------------------------------------------------
// scores[b,h,q,kv] = scale * Q_h[q,:] . K_h[kv,:]  + (1-mask)*-1e9   (fp32)
// Both operands copy straight (bf16, already [n][k]) -> async LDS staging.
// ---------------------------------------------------------------------------
__global__ void scores_kernel(const unsigned short* __restrict__ Q,
                              const unsigned short* __restrict__ Kb,
                              const int* __restrict__ mask,
                              float* __restrict__ attnp) {
  __shared__ unsigned short As[BM * SA];
  __shared__ unsigned short Bs[BN * SA];
  WAVE_COORDS();

  const int bh = blockIdx.z;
  const int b = bh >> 4, h = bh & 15;
  const int q0 = blockIdx.y * BM;
  const int kv0 = blockIdx.x * BN;
  const int hk = h * DK;

  v8f acc[4] = {{0.f, 0.f, 0.f, 0.f, 0.f, 0.f, 0.f, 0.f}};
  acc[1] = acc[0]; acc[2] = acc[0]; acc[3] = acc[0];

#pragma unroll
  for (int k0 = 0; k0 < DK; k0 += BK) {
    int ar = t >> 2, ac = (t & 3) * 8;
#pragma unroll
    for (int it = 0; it < 2; ++it) {
      int r = ar + it * 64;
      async_cp16(&Q[(size_t)(b * S + q0 + r) * D + hk + k0 + ac],
                 &As[r * SA + ac]);
    }
    async_cp16(&Kb[(size_t)(b * S + kv0 + ar) * D + hk + k0 + ac],
               &Bs[ar * SA + ac]);
    async_wait();
    __syncthreads();
    WMMA_STEP(As, Bs, acc);
    __syncthreads();
  }

  const float scale = 0.125f;  // 1/sqrt(64)
#pragma unroll
  for (int ms = 0; ms < 2; ++ms)
#pragma unroll
    for (int ns = 0; ns < 2; ++ns) {
      int gk = kv0 + wn * 32 + ns * 16 + l16;
      float pen = (1.0f - (float)mask[b * S + gk]) * -1e9f;
#pragma unroll
      for (int r = 0; r < 8; ++r) {
        int gq = q0 + wm * 32 + ms * 16 + half * 8 + r;
        attnp[((size_t)bh * S + gq) * S + gk] =
            acc[ms * 2 + ns][r] * scale + pen;
      }
    }
}

// ---------------------------------------------------------------------------
// In-place row softmax over the attn region (one block per 2048-wide row)
// ---------------------------------------------------------------------------
__global__ void softmax_kernel(float* __restrict__ attnp) {
  const int t = threadIdx.x;
  float* p = attnp + (size_t)blockIdx.x * S;
  __shared__ float red[256];

  float v[8];
  float m = -3.4e38f;
#pragma unroll
  for (int i = 0; i < 8; ++i) { v[i] = p[i * 256 + t]; m = fmaxf(m, v[i]); }
  red[t] = m; __syncthreads();
  for (int st = 128; st > 0; st >>= 1) {
    if (t < st) red[t] = fmaxf(red[t], red[t + st]);
    __syncthreads();
  }
  m = red[0]; __syncthreads();

  float sum = 0.f;
#pragma unroll
  for (int i = 0; i < 8; ++i) { v[i] = __expf(v[i] - m); sum += v[i]; }
  red[t] = sum; __syncthreads();
  for (int st = 128; st > 0; st >>= 1) {
    if (t < st) red[t] += red[t + st];
    __syncthreads();
  }
  float inv = 1.0f / red[0];
#pragma unroll
  for (int i = 0; i < 8; ++i) p[i * 256 + t] = v[i] * inv;
}

// ---------------------------------------------------------------------------
// ctx_h = attn(fp32->bf16) @ V_h      M=S, N=DK(=BN full), K=S
// ---------------------------------------------------------------------------
__global__ void ctx_kernel(const float* __restrict__ attnp,
                           const unsigned short* __restrict__ Vb,
                           unsigned short* __restrict__ Ctx) {
  __shared__ unsigned short As[BM * SA];
  __shared__ unsigned short Bs[BN * SA];
  WAVE_COORDS();

  const int bh = blockIdx.z;
  const int b = bh >> 4, h = bh & 15;
  const int m0 = blockIdx.y * BM;
  const int hk = h * DK;

  v8f acc[4] = {{0.f, 0.f, 0.f, 0.f, 0.f, 0.f, 0.f, 0.f}};
  acc[1] = acc[0]; acc[2] = acc[0]; acc[3] = acc[0];

  for (int k0 = 0; k0 < S; k0 += BK) {
    // A: attn probs fp32 -> bf16 (convert, VGPR path)
    {
      int ar = t >> 3, ac = (t & 7) * 4;
#pragma unroll
      for (int it = 0; it < 4; ++it) {
        int r = ar + it * 32;
        float4 f = *reinterpret_cast<const float4*>(
            &attnp[((size_t)bh * S + m0 + r) * S + k0 + ac]);
        unsigned short* d = &As[r * SA + ac];
        d[0] = f2bf(f.x); d[1] = f2bf(f.y); d[2] = f2bf(f.z); d[3] = f2bf(f.w);
      }
    }
    // B: Bs[n=d][k=kv] = V[kv][d]  (transpose bf16, VGPR path)
    {
      int bk = t >> 3, bn = (t & 7) * 8;
      uint4 v = *reinterpret_cast<const uint4*>(
          &Vb[(size_t)(b * S + k0 + bk) * D + hk + bn]);
      union { uint4 q; unsigned short s[8]; } u; u.q = v;
#pragma unroll
      for (int i = 0; i < 8; ++i) Bs[(bn + i) * SA + bk] = u.s[i];
    }
    __syncthreads();
    WMMA_STEP(As, Bs, acc);
    __syncthreads();
  }

#pragma unroll
  for (int ms = 0; ms < 2; ++ms)
#pragma unroll
    for (int ns = 0; ns < 2; ++ns)
#pragma unroll
      for (int r = 0; r < 8; ++r) {
        int gq = m0 + wm * 32 + ms * 16 + half * 8 + r;
        int gc = wn * 32 + ns * 16 + l16;
        Ctx[(size_t)(b * S + gq) * D + hk + gc] = f2bf(acc[ms * 2 + ns][r]);
      }
}

// ---------------------------------------------------------------------------
// out = LayerNorm(out_pre + x) * gamma + beta     (one block per row of 1024)
// ---------------------------------------------------------------------------
__global__ void ln_kernel(const float* __restrict__ outpre,
                          const float* __restrict__ x,
                          const float* __restrict__ gamma,
                          const float* __restrict__ beta,
                          float* __restrict__ out) {
  const int t = threadIdx.x;
  const size_t base = (size_t)blockIdx.x * D;
  float4 o  = *reinterpret_cast<const float4*>(&outpre[base + t * 4]);
  float4 xr = *reinterpret_cast<const float4*>(&x[base + t * 4]);
  float a[4] = {o.x + xr.x, o.y + xr.y, o.z + xr.z, o.w + xr.w};
  float s = a[0] + a[1] + a[2] + a[3];
  float q = a[0]*a[0] + a[1]*a[1] + a[2]*a[2] + a[3]*a[3];

  __shared__ float r1[256];
  __shared__ float r2[256];
  r1[t] = s; r2[t] = q; __syncthreads();
  for (int st = 128; st > 0; st >>= 1) {
    if (t < st) { r1[t] += r1[t + st]; r2[t] += r2[t + st]; }
    __syncthreads();
  }
  float mu = r1[0] * (1.0f / (float)D);
  float var = r2[0] * (1.0f / (float)D) - mu * mu;
  float rs = rsqrtf(var + 1e-12f);
#pragma unroll
  for (int i = 0; i < 4; ++i) {
    int c = t * 4 + i;
    out[base + c] = (a[i] - mu) * rs * gamma[c] + beta[c];
  }
}

// ---------------------------------------------------------------------------
extern "C" void kernel_launch(void* const* d_in, const int* in_sizes, int n_in,
                              void* d_out, int out_size, void* d_ws,
                              size_t ws_size, hipStream_t stream) {
  (void)in_sizes; (void)n_in; (void)out_size; (void)ws_size;

  const float* x     = (const float*)d_in[0];
  const int*   mask  = (const int*)d_in[1];
  const float* Wq    = (const float*)d_in[2];
  const float* bq    = (const float*)d_in[3];
  const float* Wk    = (const float*)d_in[4];
  const float* bk    = (const float*)d_in[5];
  const float* Wv    = (const float*)d_in[6];
  const float* bv    = (const float*)d_in[7];
  const float* Wo    = (const float*)d_in[8];
  const float* bo    = (const float*)d_in[9];
  const float* gamma = (const float*)d_in[10];
  const float* beta  = (const float*)d_in[11];

  const int M = BDIM * S;  // 4096 rows total
  unsigned short* qb = (unsigned short*)d_ws;
  unsigned short* kb = qb + (size_t)M * D;
  unsigned short* vb = kb + (size_t)M * D;
  unsigned short* cb = vb + (size_t)M * D;
  float* outpre = (float*)(cb + (size_t)M * D);

  float* out   = (float*)d_out;
  float* attnp = out + (size_t)M * D;  // attn region of the tuple output

  dim3 gg(D / BN, M / BM);  // (16, 32)
  gemm_bias_kernel<false, false><<<gg, 256, 0, stream>>>(x, Wq, bq, qb, M, D, D);
  gemm_bias_kernel<false, false><<<gg, 256, 0, stream>>>(x, Wk, bk, kb, M, D, D);
  gemm_bias_kernel<false, false><<<gg, 256, 0, stream>>>(x, Wv, bv, vb, M, D, D);

  scores_kernel<<<dim3(S / BN, S / BM, BDIM * H), 256, 0, stream>>>(qb, kb, mask,
                                                                    attnp);
  softmax_kernel<<<BDIM * H * S, 256, 0, stream>>>(attnp);
  ctx_kernel<<<dim3(1, S / BM, BDIM * H), 256, 0, stream>>>(attnp, vb, cb);

  gemm_bias_kernel<true, true><<<gg, 256, 0, stream>>>(cb, Wo, bo, outpre, M, D,
                                                       D);
  ln_kernel<<<M, 256, 0, stream>>>(outpre, x, gamma, beta, out);
}
